// DGM_4432406249966
// MI455X (gfx1250) — compile-verified, matrix operand order checked by reading, not run on hardware
//
#include <hip/hip_runtime.h>

typedef __attribute__((ext_vector_type(16))) _Float16 v16h;
typedef __attribute__((ext_vector_type(8)))  float    v8f;

// Problem constants (match reference)
#define BGR   16
#define NPG   2048
#define DIN   128
#define DEMB  64
#define NTOT  (BGR * NPG)        // 32768
#define GAMMA 5.0f

// d_out layout: x_aux [NTOT*DEMB], A [BGR*NPG*NPG], logprobs [BGR*NPG*NPG]
#define OFF_A  ((size_t)NTOT * DEMB)
#define OFF_LP (OFF_A + (size_t)BGR * NPG * NPG)

// ---------------- zero-fill ----------------
__global__ __launch_bounds__(256) void k_zero(float* __restrict__ p, size_t count) {
  size_t i = (size_t)blockIdx.x * blockDim.x + threadIdx.x;
  size_t stride = (size_t)gridDim.x * blockDim.x;
  for (; i < count; i += stride) p[i] = 0.0f;
}

// ---------------- embed GEMMs: h = x@Wself, xwn = x@Wnei ----------------
// One wave computes one 16x16 f32 tile via 4x v_wmma_f32_16x16x32_f16 (K=128).
__global__ __launch_bounds__(256) void k_embed_gemm(const float* __restrict__ x,
                                                    const float* __restrict__ Wself,
                                                    const float* __restrict__ Wnei,
                                                    float* __restrict__ h_out,
                                                    float* __restrict__ xwn_out) {
  const int wave = blockIdx.x * (blockDim.x >> 5) + (threadIdx.x >> 5);
  const int lane = threadIdx.x & 31;
  const int whichMat = wave >> 13;          // 16384 wave-tiles total: 0=self, 1=nei
  const int t = wave & 8191;
  const int colTile = t & 3;                // DEMB/16 = 4
  const int rowTile = t >> 2;               // NTOT/16 = 2048
  const float* __restrict__ W = whichMat ? Wnei : Wself;
  float* __restrict__ Out = whichMat ? xwn_out : h_out;

  const int hl = lane >> 4;                 // lane half (K-group select)
  const int r  = lane & 15;                 // M (A) / N (B) within tile
  const int row = rowTile * 16 + r;
  const int col = colTile * 16 + r;

  v8f c = {};
  for (int k0 = 0; k0 < DIN; k0 += 32) {
    v16h a, b;
    const float* __restrict__ arow = x + (size_t)row * DIN + k0;
    const float* __restrict__ wcol = W + (size_t)k0 * DEMB + col;
#pragma unroll
    for (int e = 0; e < 8; ++e) {
      const int k = hl * 8 + e;
      a[e]     = (_Float16)arow[k];
      a[e + 8] = (_Float16)arow[16 + k];
      b[e]     = (_Float16)wcol[(size_t)k * DEMB];
      b[e + 8] = (_Float16)wcol[(size_t)(16 + k) * DEMB];
    }
    c = __builtin_amdgcn_wmma_f32_16x16x32_f16(false, a, false, b, (short)0, c, false, false);
  }
#pragma unroll
  for (int i = 0; i < 8; ++i) {
    const int m = rowTile * 16 + hl * 8 + i;
    Out[(size_t)m * DEMB + colTile * 16 + r] = c[i];
  }
}

// ---------------- edge aggregation (scatter-add) ----------------
__global__ __launch_bounds__(256) void k_edges(const int* __restrict__ src,
                                               const int* __restrict__ dst,
                                               const float* __restrict__ xwn,
                                               float* __restrict__ agg,
                                               float* __restrict__ deg, int E) {
  const int tid = blockIdx.x * blockDim.x + threadIdx.x;
  const int e = tid >> 3;
  const int fg = tid & 7;
  if (e >= E) return;
  const int s = src[e];
  const int d = dst[e];
  const float* __restrict__ m = xwn + (size_t)s * DEMB + fg * 8;
  float* __restrict__ a = agg + (size_t)d * DEMB + fg * 8;
#pragma unroll
  for (int f = 0; f < 8; ++f) unsafeAtomicAdd(&a[f], m[f]);
  if (fg == 0) unsafeAtomicAdd(&deg[d], 1.0f);
}

// ---------------- finalize embed: x_aux = relu(h + agg/deg + b) ----------------
// wave per node; also emit f16 copy + row squared-norm.
__global__ __launch_bounds__(256) void k_finalize(float* __restrict__ x_aux,   // in: h, out: relu
                                                  const float* __restrict__ agg,
                                                  const float* __restrict__ deg,
                                                  const float* __restrict__ bvec,
                                                  _Float16* __restrict__ xb,
                                                  float* __restrict__ sq) {
  const int node = blockIdx.x * (blockDim.x >> 5) + (threadIdx.x >> 5);
  const int lane = threadIdx.x & 31;
  const float inv = 1.0f / fmaxf(deg[node], 1.0f);
  float ss = 0.0f;
#pragma unroll
  for (int f = lane; f < DEMB; f += 32) {
    const size_t idx = (size_t)node * DEMB + f;
    float v = x_aux[idx] + agg[idx] * inv + bvec[f];
    v = fmaxf(v, 0.0f);
    x_aux[idx] = v;
    xb[idx] = (_Float16)v;
    ss += v * v;
  }
#pragma unroll
  for (int o = 16; o; o >>= 1) ss += __shfl_xor(ss, o, 32);
  if (lane == 0) sq[node] = ss;
}

// ---------------- fused Gram GEMM + distance + layernorm + entmax15 ----------------
// One block (512 thr = 16 waves) owns a 16-row strip of one graph:
//   Phase 1: wave w computes col tiles jt = w*8..w*8+7 via WMMA (A tile cached in
//            registers across jt), fuses -sqrt(d2)+noise, stages strip in LDS (128 KB).
//   Phase 2: wave w owns local row w; its 2048 elements live in 64 regs/lane;
//            layernorm + tau bisection use only wave shfl reductions; writes A + logP.
__global__ __launch_bounds__(512) void k_dist_entmax(const _Float16* __restrict__ xb,
                                                     const float* __restrict__ sq,
                                                     const float* __restrict__ noise,
                                                     float* __restrict__ Aout,
                                                     float* __restrict__ LPout) {
  extern __shared__ float lds[];            // 16 * NPG floats = 128 KB
  const int wave = threadIdx.x >> 5;        // 0..15
  const int lane = threadIdx.x & 31;
  const int g  = blockIdx.x >> 7;           // 128 strips per graph
  const int it = blockIdx.x & 127;
  const int hl = lane >> 4;
  const int r  = lane & 15;
  const int iBase = it * 16;

  const _Float16* __restrict__ xg = xb + (size_t)g * NPG * DEMB;
  const float* __restrict__ sqg = sq + (size_t)g * NPG;
  const size_t gbase = (size_t)g * NPG * NPG;

  // ---- Phase 1: Gram strip -> logits in LDS ----
  // Preload the strip's A tile (both K-steps) once per wave.
  v16h a0, a1;
  {
    const _Float16* __restrict__ arow = xg + (size_t)(iBase + r) * DEMB;
#pragma unroll
    for (int e = 0; e < 8; ++e) {
      const int k = hl * 8 + e;
      a0[e]     = arow[k];        a0[e + 8] = arow[16 + k];
      a1[e]     = arow[32 + k];   a1[e + 8] = arow[48 + k];
    }
  }
  float sqA[8];
#pragma unroll
  for (int i = 0; i < 8; ++i) sqA[i] = sqg[iBase + hl * 8 + i];

  for (int jj = 0; jj < 8; ++jj) {
    const int jt = wave * 8 + jj;
    const int ncol = jt * 16 + r;
    v16h b0, b1;
    const _Float16* __restrict__ brow = xg + (size_t)ncol * DEMB;
#pragma unroll
    for (int e = 0; e < 8; ++e) {
      const int k = hl * 8 + e;
      b0[e]     = brow[k];        b0[e + 8] = brow[16 + k];
      b1[e]     = brow[32 + k];   b1[e + 8] = brow[48 + k];
    }
    v8f c = {};
    c = __builtin_amdgcn_wmma_f32_16x16x32_f16(false, a0, false, b0, (short)0, c, false, false);
    c = __builtin_amdgcn_wmma_f32_16x16x32_f16(false, a1, false, b1, (short)0, c, false, false);
    const float sqn = sqg[ncol];
#pragma unroll
    for (int i = 0; i < 8; ++i) {
      const int m = hl * 8 + i;
      const float d2 = sqA[i] + sqn - 2.0f * c[i];
      const float D = sqrtf(fmaxf(d2, 0.0f) + 1e-12f);
      lds[m * NPG + ncol] = noise[gbase + (size_t)(iBase + m) * NPG + ncol] - D;
    }
  }
  __syncthreads();

  // ---- Phase 2: one wave per row; 64 elements per lane in registers ----
  const int m = wave;
  float z[64];
  float s1 = 0.0f, s2 = 0.0f;
#pragma unroll
  for (int q = 0; q < 64; ++q) {
    z[q] = lds[m * NPG + lane + q * 32];
    s1 += z[q];
    s2 += z[q] * z[q];
  }
#pragma unroll
  for (int o = 16; o; o >>= 1) {
    s1 += __shfl_xor(s1, o, 32);
    s2 += __shfl_xor(s2, o, 32);
  }
  const float mu = s1 * (1.0f / NPG);
  const float sd = sqrtf(fmaxf(s2 * (1.0f / NPG) - mu * mu, 0.0f));
  const float scale = 0.5f * GAMMA / (sd + 1e-5f);   // 0.5 from entmax core

  float mx = -3.4e38f;
#pragma unroll
  for (int q = 0; q < 64; ++q) {
    z[q] = (z[q] - mu) * scale;
    mx = fmaxf(mx, z[q]);
  }
#pragma unroll
  for (int o = 16; o; o >>= 1) mx = fmaxf(mx, __shfl_xor(mx, o, 32));

  // bisection on f(tau) = sum(max(z - tau, 0)^2) = 1; f(mx-1) >= 1, f(mx) = 0
  float lo = mx - 1.0f, hi = mx;
  for (int itr = 0; itr < 32; ++itr) {
    const float tau = 0.5f * (lo + hi);
    float f = 0.0f;
#pragma unroll
    for (int q = 0; q < 64; ++q) {
      const float d = z[q] - tau;
      f += (d > 0.0f) ? d * d : 0.0f;
    }
#pragma unroll
    for (int o = 16; o; o >>= 1) f += __shfl_xor(f, o, 32);
    if (f >= 1.0f) lo = tau; else hi = tau;
  }
  const float tau = 0.5f * (lo + hi);

  const size_t rowbase = gbase + (size_t)(iBase + m) * NPG;
#pragma unroll
  for (int q = 0; q < 64; ++q) {
    const float d = fmaxf(z[q] - tau, 0.0f);
    const float p = d * d;
    Aout[rowbase + lane + q * 32]  = p;
    LPout[rowbase + lane + q * 32] = logf(p + 1e-6f);
  }
}

extern "C" void kernel_launch(void* const* d_in, const int* in_sizes, int n_in,
                              void* d_out, int out_size, void* d_ws, size_t ws_size,
                              hipStream_t stream) {
  (void)n_in; (void)out_size; (void)ws_size;
  const float* x     = (const float*)d_in[0];
  const int*   eidx  = (const int*)d_in[1];
  const float* Wself = (const float*)d_in[4];
  const float* Wnei  = (const float*)d_in[5];
  const float* bvec  = (const float*)d_in[6];
  const float* noise = (const float*)d_in[7];
  const int E = in_sizes[1] / 2;
  const int* src = eidx;
  const int* dst = eidx + E;

  float* out   = (float*)d_out;
  float* x_aux = out;                 // h staged here, relu'd in place
  float* Areg  = out + OFF_A;
  float* LPreg = out + OFF_LP;

  // workspace layout (bytes): xwn | agg | deg | sq | xb(f16)
  char* ws = (char*)d_ws;
  float*    xwn = (float*)(ws);                        // NTOT*DEMB f32 = 8 MB
  float*    agg = (float*)(ws + 8388608);              // NTOT*DEMB f32 = 8 MB
  float*    deg = (float*)(ws + 16777216);             // NTOT f32
  float*    sq  = (float*)(ws + 16908288);             // NTOT f32
  _Float16* xb  = (_Float16*)(ws + 17039360);          // NTOT*DEMB f16 = 4 MB

  // 1) zero agg + deg (contiguous)
  k_zero<<<1024, 256, 0, stream>>>(agg, (size_t)NTOT * DEMB + NTOT);
  // 2) embed GEMMs (WMMA)
  k_embed_gemm<<<2048, 256, 0, stream>>>(x, Wself, Wnei, x_aux, xwn);
  // 3) edge scatter-add
  k_edges<<<(E * 8 + 255) / 256, 256, 0, stream>>>(src, dst, xwn, agg, deg, E);
  // 4) relu fuse + f16 copy + sq norms
  k_finalize<<<NTOT / 8, 256, 0, stream>>>(x_aux, agg, deg, bvec, xb, sq);
  // 5) fused Gram GEMM (WMMA) + distance/noise + layernorm + entmax15 + logprobs
  k_dist_entmax<<<BGR * (NPG / 16), 512, 16 * NPG * sizeof(float), stream>>>(
      xb, sq, noise, Areg, LPreg);
}
